// VectorNet_46059229283187
// MI455X (gfx1250) — matrix operand order repaired and checked by reading, not compiled, not run on hardware
//
#include <hip/hip_runtime.h>
#include <hip/hip_bf16.h>
#include <math.h>

typedef __attribute__((ext_vector_type(16))) _Float16 v16h;
typedef __attribute__((ext_vector_type(8)))  float    v8f;

#define NB    64    // batch
#define NAG   49    // agent nodes
#define NMP   19    // map nodes
#define NPLY  128   // map polylines P
#define IND   64    // input feature dim
#define HD    64    // hidden H
#define H2D   128   // 2H
#define NNODE 129   // 1 + P GAT nodes
#define OUTD  60

#define XPAD  (H2D + 8)   // f16 row stride: 272B, multiple of 16B, de-conflicted
#define WPAD  (H2D + 8)   // f16 K-stride for transposed weight tile
#define YPAD  (HD + 1)    // f32 row stride 260B -> bank = (r+c) mod 64

// ---------------- WMMA fragment helpers (CDNA5 wave32 layouts) ----------------
// A 16x32 f16: lane l<16 -> row m=l, halves 0-15 = K k0..k0+15
//              lane l>=16 -> row m=l-16, halves 0-15 = K k0+16..k0+31
__device__ __forceinline__ v16h ld_a(const _Float16* x, int ldw, int m0, int k0, int lane) {
  const _Float16* p = x + (m0 + (lane & 15)) * ldw + k0 + ((lane & 16) ? 16 : 0);
  v16h a;
#pragma unroll
  for (int j = 0; j < 16; ++j) a[j] = p[j];
  return a;
}
// B 32x16 f16 from TRANSPOSED tile wt[n][k] (ldk stride): lane -> col n, 16 contiguous K
__device__ __forceinline__ v16h ld_bt(const _Float16* wt, int ldk, int k0, int n0, int lane) {
  const _Float16* p = wt + (n0 + (lane & 15)) * ldk + k0 + ((lane & 16) ? 16 : 0);
  v16h b;
#pragma unroll
  for (int j = 0; j < 16; ++j) b[j] = p[j];
  return b;
}
// D 16x16 f32: lane l<16 -> col l, vgpr r -> row r; lane>=16 -> col l-16, row r+8
__device__ __forceinline__ void st_d(float* y, int ldw, int m0, int n0, int lane, v8f acc, int mlim) {
  int c  = n0 + (lane & 15);
  int rb = m0 + ((lane & 16) ? 8 : 0);
#pragma unroll
  for (int r = 0; r < 8; ++r) { int rr = rb + r; if (rr < mlim) y[rr * ldw + c] = acc[r]; }
}

struct SubnetParams { const float *w[3], *b[3], *g[3], *bt[3]; };

// ---------------- SubNetwork: 3x (Linear + LN + ReLU -> concat(h, neighbor_max)) ----
// NGR graphs packed tightly (graph g at rows [g*nn, g*nn+nn)), padded to MP rows.
// 128 threads = 4 wave32 waves, one 16-col n-tile per wave.
template<int MP, int NGR>
__global__ void __launch_bounds__(128)
subnet_kernel(const float* __restrict__ xin, int nn, SubnetParams prm,
              float* __restrict__ feats, int is_map)
{
  __shared__ _Float16 Xs[MP][XPAD];     // activations, layer0 uses cols 0..63
  __shared__ _Float16 Wt[HD][WPAD];     // weight tile TRANSPOSED: Wt[n][k]
  __shared__ float    Ys[MP][YPAD];     // f32 GEMM result / post-LN h
  __shared__ float    m1s[NGR][HD], m2s[NGR][HD];
  __shared__ float    bbs[HD], ggs[HD], bts[HD];

  const int tid = threadIdx.x, lane = tid & 31, wave = tid >> 5;
  const int NT = NGR * nn;                       // real rows in this block
  const int G0 = blockIdx.x * NGR;               // first graph index

  if (tid == 0) {                                // gfx1250 global_prefetch_b8
    __builtin_prefetch(prm.w[0], 0, 3);
    __builtin_prefetch(prm.w[1], 0, 3);
    __builtin_prefetch(prm.w[2], 0, 3);
  }

  for (int idx = tid; idx < MP * IND; idx += 128) {
    int r = idx / IND, c = idx % IND;
    _Float16 v = (_Float16)0.f;
    if (r < NT) {
      int g = r / nn, rr = r % nn;
      v = (_Float16)xin[((size_t)(G0 + g) * nn + rr) * IND + c];
    }
    Xs[r][c] = v;
  }
  __syncthreads();

#pragma unroll
  for (int l = 0; l < 3; ++l) {
    const int K = (l == 0) ? IND : H2D;
    const float* Wg = prm.w[l];
    for (int idx = tid; idx < K * HD; idx += 128) {     // store transposed
      int k = idx / HD, n = idx % HD;
      Wt[n][k] = (_Float16)Wg[idx];
    }
    if (tid < HD) { bbs[tid] = prm.b[l][tid]; ggs[tid] = prm.g[l][tid]; bts[tid] = prm.bt[l][tid]; }
    __syncthreads();

    // Y[MP x 64] = X[MP x K] @ W[K x 64]
    {
      const int nt = wave;
#pragma unroll
      for (int mt = 0; mt < MP / 16; ++mt) {
        v8f acc = {};
        for (int ks = 0; ks < K; ks += 32) {
          v16h a = ld_a (&Xs[0][0], XPAD, mt * 16, ks, lane);
          v16h b = ld_bt(&Wt[0][0], WPAD, ks, nt * 16, lane);
          acc = __builtin_amdgcn_wmma_f32_16x16x32_f16(false, a, false, b,
                                                       (short)0, acc, false, false);
        }
        st_d(&Ys[0][0], YPAD, mt * 16, nt * 16, lane, acc, MP);
      }
    }
    __syncthreads();

    // bias + LayerNorm + ReLU, per row
    if (tid < MP) {
      const int r = tid;
      float mu = 0.f;
      for (int c = 0; c < HD; ++c) mu += Ys[r][c] + bbs[c];
      mu *= (1.f / HD);
      float var = 0.f;
      for (int c = 0; c < HD; ++c) { float t = Ys[r][c] + bbs[c] - mu; var += t * t; }
      var *= (1.f / HD);
      float rs = rsqrtf(var + 1e-5f);
      for (int c = 0; c < HD; ++c) {
        float h = ggs[c] * (Ys[r][c] + bbs[c] - mu) * rs + bts[c];
        Ys[r][c] = fmaxf(h, 0.f);
      }
    }
    __syncthreads();

    // per-graph per-channel top-2 over real nodes (neighbor-max trick)
    for (int idx = tid; idx < NGR * HD; idx += 128) {
      int g = idx / HD, c = idx % HD;
      float a1 = -INFINITY, a2 = -INFINITY;
      for (int r = g * nn; r < g * nn + nn; ++r) {
        float v = Ys[r][c];
        if (v > a1) { a2 = a1; a1 = v; } else if (v > a2) { a2 = v; }
      }
      m1s[g][c] = a1; m2s[g][c] = a2;
    }
    __syncthreads();

    // X_next = concat(h, where(h==max, second, max))  (f16)
    for (int idx = tid; idx < MP * HD; idx += 128) {
      int r = idx / HD, c = idx % HD;
      if (r < NT) {
        int g = r / nn;
        float h  = Ys[r][c];
        float nb = (h == m1s[g][c]) ? m2s[g][c] : m1s[g][c];
        Xs[r][c]      = (_Float16)h;
        Xs[r][HD + c] = (_Float16)nb;
      } else {
        Xs[r][c]      = (_Float16)0.f;
        Xs[r][HD + c] = (_Float16)0.f;
      }
    }
    __syncthreads();
  }

  // per-graph node max-pool -> feats row (node 0 = agent, 1+p = map polyline)
  for (int idx = tid; idx < NGR * H2D; idx += 128) {
    int g = idx / H2D, c = idx % H2D;
    float m = -INFINITY;
    for (int r = g * nn; r < g * nn + nn; ++r) m = fmaxf(m, (float)Xs[r][c]);
    int gi = G0 + g;
    float* dst;
    if (is_map) {
      int p = gi / NB, b = gi % NB;
      dst = feats + ((size_t)b * NNODE + 1 + p) * H2D;
    } else {
      dst = feats + (size_t)gi * NNODE * H2D;
    }
    dst[c] = m;
  }
}

// ---------------- GAT stage 1: Z = feats @ gat_fc_w, per-batch WMMA ----------------
__global__ void __launch_bounds__(128)
gat_gemm_kernel(const float* __restrict__ feats, const float* __restrict__ fcw,
                float* __restrict__ Z)
{
  __shared__ _Float16 Xs[144][XPAD];                // 129 nodes padded to 144
  const int tid = threadIdx.x, lane = tid & 31, wave = tid >> 5;
  const float* fg = feats + (size_t)blockIdx.x * NNODE * H2D;

  if (tid == 0) __builtin_prefetch(fcw, 0, 3);

  for (int idx = tid; idx < 144 * H2D; idx += 128) {
    int r = idx / H2D, c = idx % H2D;
    Xs[r][c] = (r < NNODE) ? (_Float16)fg[r * H2D + c] : (_Float16)0.f;
  }
  __syncthreads();

  float* zg = Z + (size_t)blockIdx.x * NNODE * H2D;
  for (int t = wave; t < 8; t += 4) {               // 8 n-tiles, 2 per wave (uniform)
    v16h bfr[4];
#pragma unroll
    for (int ks = 0; ks < 4; ++ks) {                // build B frags straight from global f32
      int c  = t * 16 + (lane & 15);
      int kb = ks * 32 + ((lane & 16) ? 16 : 0);
#pragma unroll
      for (int j = 0; j < 16; ++j) bfr[ks][j] = (_Float16)fcw[(kb + j) * H2D + c];
    }
    for (int mt = 0; mt < 9; ++mt) {
      v8f acc = {};
#pragma unroll
      for (int ks = 0; ks < 4; ++ks) {
        v16h a = ld_a(&Xs[0][0], XPAD, mt * 16, ks * 32, lane);
        acc = __builtin_amdgcn_wmma_f32_16x16x32_f16(false, a, false, bfr[ks],
                                                     (short)0, acc, false, false);
      }
      int cc = t * 16 + (lane & 15);
      int rb = mt * 16 + ((lane & 16) ? 8 : 0);
#pragma unroll
      for (int r = 0; r < 8; ++r) {
        int rr = rb + r;
        if (rr < NNODE) zg[(size_t)rr * H2D + cc] = acc[r];
      }
    }
  }
}

// ---------------- GAT stage 2: attention softmax at dst node 0 + MLP readout -------
__global__ void __launch_bounds__(128)
gat_attn_kernel(const float* __restrict__ Z, const float* __restrict__ attw,
                const float* __restrict__ mlpw, const float* __restrict__ mlpb,
                float* __restrict__ out)
{
  __shared__ float sb[NNODE], eb[H2D], hg[H2D], red[4];
  const int tid = threadIdx.x;
  const float* zg = Z + (size_t)blockIdx.x * NNODE * H2D;

  for (int i = tid; i < NNODE; i += 128) {          // source scores s_i
    float s = 0.f;
    for (int c = 0; c < H2D; ++c) s += zg[(size_t)i * H2D + c] * attw[c];
    sb[i] = s;
  }
  if (tid == 0) {                                   // destination score d_0
    float d0 = 0.f;
    for (int c = 0; c < H2D; ++c) d0 += zg[c] * attw[H2D + c];
    red[0] = d0;
  }
  __syncthreads();
  {                                                 // e_i = leaky_relu(s_i + d_0), i=1..128
    float e = sb[tid + 1] + red[0];
    eb[tid] = (e < 0.f) ? 0.01f * e : e;
  }
  __syncthreads();
  if (tid == 0) { float m = -INFINITY; for (int i = 0; i < H2D; ++i) m = fmaxf(m, eb[i]); red[1] = m; }
  __syncthreads();
  eb[tid] = __expf(eb[tid] - red[1]);
  __syncthreads();
  if (tid == 0) { float s = 0.f; for (int i = 0; i < H2D; ++i) s += eb[i]; red[2] = s; }
  __syncthreads();
  {                                                 // hg0 = sum_i alpha_i * z_i
    float inv = 1.f / red[2], acc = 0.f;
    for (int i = 0; i < H2D; ++i) acc += eb[i] * zg[(size_t)(i + 1) * H2D + tid];
    hg[tid] = acc * inv;
  }
  __syncthreads();
  if (tid < OUTD) {                                 // out = hg0 @ mlp_w + mlp_b
    float acc = mlpb[tid];
    for (int c = 0; c < H2D; ++c) acc += hg[c] * mlpw[c * OUTD + tid];
    out[blockIdx.x * OUTD + tid] = acc;
  }
}

extern "C" void kernel_launch(void* const* d_in, const int* in_sizes, int n_in,
                              void* d_out, int out_size, void* d_ws, size_t ws_size,
                              hipStream_t stream) {
  (void)in_sizes; (void)n_in; (void)out_size; (void)ws_size;
  const float* agent = (const float*)d_in[0];
  const float* mapf  = (const float*)d_in[1];
  // d_in[2] = map_mask (all ones, static -> ignored, as in reference)
  SubnetParams ap, mp;
  for (int i = 0; i < 3; ++i) {
    ap.w[i]  = (const float*)d_in[3  + 4 * i];
    ap.b[i]  = (const float*)d_in[4  + 4 * i];
    ap.g[i]  = (const float*)d_in[5  + 4 * i];
    ap.bt[i] = (const float*)d_in[6  + 4 * i];
    mp.w[i]  = (const float*)d_in[15 + 4 * i];
    mp.b[i]  = (const float*)d_in[16 + 4 * i];
    mp.g[i]  = (const float*)d_in[17 + 4 * i];
    mp.bt[i] = (const float*)d_in[18 + 4 * i];
  }
  const float* fcw  = (const float*)d_in[27];
  const float* attw = (const float*)d_in[28];
  const float* mlpw = (const float*)d_in[29];
  const float* mlpb = (const float*)d_in[30];

  float* feats = (float*)d_ws;                               // [B,129,128]
  float* Z     = feats + (size_t)NB * NNODE * H2D;           // [B,129,128]

  // agent: 1 graph (49 nodes -> 64 rows) per block, 64 blocks
  subnet_kernel<64, 1><<<NB, 128, 0, stream>>>(agent, NAG, ap, feats, 0);
  // map: 2 graphs tight-packed (38 nodes -> 48 rows) per block, 4096 blocks
  subnet_kernel<48, 2><<<(NPLY * NB) / 2, 128, 0, stream>>>(mapf, NMP, mp, feats, 1);
  gat_gemm_kernel<<<NB, 128, 0, stream>>>(feats, fcw, Z);
  gat_attn_kernel<<<NB, 128, 0, stream>>>(Z, attw, mlpw, mlpb, (float*)d_out);
}
